// MVN_DDI_4483945857414
// MI455X (gfx1250) — compile-verified, matrix operand order checked by reading, not run on hardware
//
#include <hip/hip_runtime.h>
#include <hip/hip_bf16.h>

// ---------------- problem constants (match reference) ----------------
constexpr int kB    = 512;          // drug pairs
constexpr int kNPG  = 32;           // nodes per graph
constexpr int kEPG  = 64;           // edges per graph
constexpr int kNF   = 55;
constexpr int kEF   = 13;
constexpr int kD    = 128;
constexpr int kDE   = 64;
constexpr int kNBLK = 4;
constexpr int kN    = kB * kNPG;    // 16384
constexpr int kE    = kB * kEPG;    // 32768

typedef __attribute__((ext_vector_type(16))) _Float16 v16h;
typedef __attribute__((ext_vector_type(8)))  float    v8f;

// ---------------- WMMA GEMM: C[M,Nc] = A[M,K] @ W[K,Nc] (+bias)(+relu) ----------
// Block = 8 waves covering 8 m-tiles x one 64-wide column block. W is staged once
// per block into LDS as f16, pre-swizzled into the WMMA B-fragment layout
// ([ktile][ntile][lane][16 halfs]) so each wave reads a fragment with aligned
// 32-byte LDS loads. Per K-step: load A fragment + ALL four B fragments first,
// then issue 4 back-to-back v_wmma_f32_16x16x32_f16 (dscnt waits overlap).
template <int K>
__global__ void k_gemm_wmma(const float* __restrict__ A, const float* __restrict__ W,
                            const float* __restrict__ bias, float* __restrict__ C,
                            int M, int Nc, int act) {
  constexpr int KT = K / 32;                       // K-steps
  __shared__ _Float16 sw[KT * 4 * 32 * 16];        // B fragments, 8KB/16KB
  int tcb = blockIdx.x;                            // 64-wide column block
  int tm0 = blockIdx.y * 8;                        // first m-tile of block
  int tid = threadIdx.x;                           // blockDim = 256

  // ---- cooperative stage: W (f32, strided) -> LDS f16 in fragment layout ----
  for (int ent = tid; ent < KT * 128; ent += 256) {
    int kt   = ent >> 7;
    int rem  = ent & 127;
    int t    = rem >> 5;
    int lane = rem & 31;
    int col   = tcb * 64 + t * 16 + (lane & 15);
    int kbase = kt * 32 + (lane >> 4) * 8;
    _Float16* dst = sw + ent * 16;
    const float* src = W + (size_t)kbase * Nc + col;
#pragma unroll
    for (int j = 0; j < 8; ++j) {
      dst[j]     = (_Float16)src[(size_t)j * Nc];
      dst[j + 8] = (_Float16)src[(size_t)(16 + j) * Nc];
    }
  }
  __syncthreads();

  int wv = tid >> 5;                 // wave id -> m-tile offset
  int tm = tm0 + wv;
  if (tm >= (M >> 4)) return;        // wave-uniform
  int lane = tid & 31;
  int hsel = lane >> 4;              // K sub-offset select (0/8)
  int lr   = lane & 15;
  v8f acc[4] = {};
  const float* arow = A + (size_t)(tm * 16 + lr) * K + hsel * 8;
#pragma unroll
  for (int kt = 0; kt < KT; ++kt) {
    // A fragment: 16 f16 per lane from 4 aligned float4 loads
    v16h a;
    const float4* a4 = (const float4*)(arow + kt * 32);
    float4 x0 = a4[0], x1 = a4[1], x2 = a4[4], x3 = a4[5];
    a[0]  = (_Float16)x0.x; a[1]  = (_Float16)x0.y; a[2]  = (_Float16)x0.z; a[3]  = (_Float16)x0.w;
    a[4]  = (_Float16)x1.x; a[5]  = (_Float16)x1.y; a[6]  = (_Float16)x1.z; a[7]  = (_Float16)x1.w;
    a[8]  = (_Float16)x2.x; a[9]  = (_Float16)x2.y; a[10] = (_Float16)x2.z; a[11] = (_Float16)x2.w;
    a[12] = (_Float16)x3.x; a[13] = (_Float16)x3.y; a[14] = (_Float16)x3.z; a[15] = (_Float16)x3.w;
    // fetch all four B fragments before the WMMA burst
    v16h bfr[4];
#pragma unroll
    for (int t = 0; t < 4; ++t)
      bfr[t] = *(const v16h*)(sw + (((kt * 4 + t) * 32) + lane) * 16);
#pragma unroll
    for (int t = 0; t < 4; ++t)
      acc[t] = __builtin_amdgcn_wmma_f32_16x16x32_f16(false, a, false, bfr[t],
                                                      (short)0, acc[t], false, false);
  }
#pragma unroll
  for (int t = 0; t < 4; ++t) {
    int col = tcb * 64 + t * 16 + lr;
    float bv = bias ? bias[col] : 0.0f;
#pragma unroll
    for (int r = 0; r < 8; ++r) {
      int row = tm * 16 + r + hsel * 8;
      float v = acc[t][r] + bv;
      if (act == 1) v = fmaxf(v, 0.0f);
      C[(size_t)row * Nc + col] = v;
    }
  }
}

// ---------------- init projections ----------------
__global__ void k_node_proj(const float* __restrict__ X, const float* __restrict__ W,
                            const float* __restrict__ b, float* __restrict__ Y) {
  __shared__ float sx[kNF];
  int n = blockIdx.x, c = threadIdx.x;   // blockDim = 128
  if (c < kNF) sx[c] = X[(size_t)n * kNF + c];
  __syncthreads();
  float acc = b[c];
  for (int k = 0; k < kNF; ++k) acc += sx[k] * W[k * kD + c];
  Y[(size_t)n * kD + c] = acc;
}

__global__ void k_edge_proj(const float* __restrict__ X, const float* __restrict__ W,
                            const float* __restrict__ b, float* __restrict__ Y) {
  __shared__ float sx[kEF];
  int e = blockIdx.x, c = threadIdx.x;   // blockDim = 64
  if (c < kEF) sx[c] = X[(size_t)e * kEF + c];
  __syncthreads();
  float acc = b[c];
  for (int k = 0; k < kEF; ++k) acc += sx[k] * W[k * kDE + c];
  Y[(size_t)e * kDE + c] = fmaxf(acc, 0.0f);
}

// ---------------- graph LayerNorm (PyG mode='graph') + optional ReLU ------------
__global__ void k_graph_ln(const float* __restrict__ X, float* __restrict__ Y,
                           const float* __restrict__ w, const float* __restrict__ b,
                           int relu) {
  __shared__ float rs[256], rq[256];
  int g = blockIdx.x, tid = threadIdx.x; // blockDim = 256
  const float* xg = X + (size_t)g * kNPG * kD;
  float s = 0.f, q = 0.f;
  for (int i = tid; i < kNPG * kD; i += 256) { float v = xg[i]; s += v; q += v * v; }
  rs[tid] = s; rq[tid] = q; __syncthreads();
  for (int st = 128; st > 0; st >>= 1) {
    if (tid < st) { rs[tid] += rs[tid + st]; rq[tid] += rq[tid + st]; }
    __syncthreads();
  }
  const float inv_cnt = 1.0f / (kNPG * kD);
  float mean = rs[0] * inv_cnt;
  float var  = rq[0] * inv_cnt - mean * mean;
  float invs = rsqrtf(fmaxf(var, 0.0f) + 1e-5f);
  float* yg = Y + (size_t)g * kNPG * kD;
  for (int i = tid; i < kNPG * kD; i += 256) {
    int c = i & (kD - 1);
    float v = (xg[i] - mean) * invs * w[c] + b[c];
    if (relu) v = fmaxf(v, 0.0f);
    yg[i] = v;
  }
}

// ---------------- elementwise ----------------
__global__ void k_elu(const float* __restrict__ X, float* __restrict__ Y, int n) {
  int i = blockIdx.x * blockDim.x + threadIdx.x;
  if (i < n) { float v = X[i]; Y[i] = v > 0.f ? v : (expf(v) - 1.0f); }
}
__global__ void k_relu(const float* __restrict__ X, float* __restrict__ Y, int n) {
  int i = blockIdx.x * blockDim.x + threadIdx.x;
  if (i < n) Y[i] = fmaxf(X[i], 0.0f);
}

// ---------------- per-graph GAT (conv with edge-attrs, or intra) ----------------
// One workgroup per graph; everything in LDS (32 nodes, 64 edges).
__global__ void k_gat_graph(const float* __restrict__ hsM, const float* __restrict__ ewM,
                            const float* __restrict__ att_src, const float* __restrict__ att_dst,
                            const float* __restrict__ att_e,
                            const int* __restrict__ srcI, const int* __restrict__ dstI,
                            const float* __restrict__ bias, float* __restrict__ out,
                            int ldo, int col0, int H, int C) {
  __shared__ float shs[kNPG * 128];
  __shared__ float slog[kEPG * 2], salpha[kEPG * 2];
  __shared__ float ss[kNPG * 2], sd[kNPG * 2], smax[kNPG * 2], ssum[kNPG * 2];
  __shared__ int ssl[kEPG], sdl[kEPG];
  int g = blockIdx.x, tid = threadIdx.x, bdim = blockDim.x;
  int HC = H * C;
  for (int i = tid; i < kNPG * HC; i += bdim) shs[i] = hsM[(size_t)g * kNPG * HC + i];
  if (tid < kEPG) {
    ssl[tid] = srcI[g * kEPG + tid] - g * kNPG;
    sdl[tid] = dstI[g * kEPG + tid] - g * kNPG;
  }
  __syncthreads();
  if (tid < kNPG * H) {
    int n = tid / H, h = tid % H;
    float a = 0.f, d = 0.f;
    for (int c = 0; c < C; ++c) {
      float v = shs[n * HC + h * C + c];
      a += v * att_src[h * C + c];
      d += v * att_dst[h * C + c];
    }
    ss[tid] = a; sd[tid] = d;
  }
  __syncthreads();
  for (int i = tid; i < kEPG * H; i += bdim) {
    int e = i / H, h = i % H;
    float l = ss[ssl[e] * H + h] + sd[sdl[e] * H + h];
    if (ewM) {
      const float* ep = ewM + (size_t)(g * kEPG + e) * HC + h * C;
      for (int c = 0; c < C; ++c) l += ep[c] * att_e[h * C + c];
    }
    slog[i] = l > 0.f ? l : 0.2f * l;   // leaky_relu(0.2)
  }
  __syncthreads();
  if (tid < kNPG * H) {
    int n = tid / H, h = tid % H;
    float m = -1e30f;
    for (int e = 0; e < kEPG; ++e) if (sdl[e] == n) m = fmaxf(m, slog[e * H + h]);
    float s = 0.f;
    for (int e = 0; e < kEPG; ++e) if (sdl[e] == n) s += expf(slog[e * H + h] - m);
    smax[tid] = m; ssum[tid] = s;
  }
  __syncthreads();
  for (int i = tid; i < kEPG * H; i += bdim) {
    int e = i / H, h = i % H;
    int n = sdl[e];
    salpha[i] = expf(slog[i] - smax[n * H + h]) / (ssum[n * H + h] + 1e-16f);
  }
  __syncthreads();
  for (int i = tid; i < kNPG * HC; i += bdim) {
    int n = i / HC, hc = i % HC, h = hc / C;
    float acc = 0.f;
    for (int e = 0; e < kEPG; ++e)
      if (sdl[e] == n) acc += salpha[e * H + h] * shs[ssl[e] * HC + hc];
    out[(size_t)(g * kNPG + n) * ldo + col0 + hc] = acc + bias[hc];
  }
}

// ---------------- dense inter-graph GAT (bipartite full 32x32 per pair) --------
// blockIdx.y==0: t_inter (src = heads via Wsrc, dst = tails via Wdst, out -> repT)
// blockIdx.y==1: h_inter (src = tails via Wsrc, dst = heads via Wdst, out -> repH)
__global__ void k_inter_gat(const float* __restrict__ qh, const float* __restrict__ kdt,
                            const float* __restrict__ qt, const float* __restrict__ kdh,
                            const float* __restrict__ att_src, const float* __restrict__ att_dst,
                            const float* __restrict__ bias,
                            float* __restrict__ repH, float* __restrict__ repT,
                            int ldo, int col0) {
  const int H = 2, C = 32, HC = 64;
  __shared__ float ssrc[kNPG * 64], salpha[kNPG * 64];
  __shared__ float ss[kNPG * 2], sd[kNPG * 2], smax[kNPG * 2], ssum[kNPG * 2];
  int g = blockIdx.x, dir = blockIdx.y, tid = threadIdx.x, bdim = blockDim.x;
  const float* hsrc = (dir == 0) ? qh : qt;
  const float* hdst = (dir == 0) ? kdt : kdh;
  float* out = (dir == 0) ? repT : repH;
  for (int i = tid; i < kNPG * HC; i += bdim) ssrc[i] = hsrc[(size_t)g * kNPG * HC + i];
  __syncthreads();
  if (tid < kNPG * H) {
    int n = tid / H, h = tid % H;
    float a = 0.f;
    for (int c = 0; c < C; ++c) a += ssrc[n * HC + h * C + c] * att_src[h * C + c];
    ss[tid] = a;
    float d = 0.f;
    const float* hp = hdst + (size_t)(g * kNPG + n) * HC + h * C;
    for (int c = 0; c < C; ++c) d += hp[c] * att_dst[h * C + c];
    sd[tid] = d;
  }
  __syncthreads();
  if (tid < kNPG * H) {          // per (dstN,h): softmax over 32 src nodes
    int h = tid % H;
    float dterm = sd[tid];
    float m = -1e30f;
    for (int s2 = 0; s2 < kNPG; ++s2) {
      float l = ss[s2 * H + h] + dterm; l = l > 0.f ? l : 0.2f * l;
      m = fmaxf(m, l);
    }
    float sum = 0.f;
    for (int s2 = 0; s2 < kNPG; ++s2) {
      float l = ss[s2 * H + h] + dterm; l = l > 0.f ? l : 0.2f * l;
      sum += expf(l - m);
    }
    smax[tid] = m; ssum[tid] = sum;
  }
  __syncthreads();
  for (int i = tid; i < kNPG * kNPG * H; i += bdim) {   // 2048 alphas
    int dn = i / (kNPG * H); int rem = i % (kNPG * H);
    int h = rem / kNPG; int sn = rem % kNPG;
    float l = ss[sn * H + h] + sd[dn * H + h]; l = l > 0.f ? l : 0.2f * l;
    salpha[dn * 64 + h * 32 + sn] =
        expf(l - smax[dn * H + h]) / (ssum[dn * H + h] + 1e-16f);
  }
  __syncthreads();
  for (int i = tid; i < kNPG * HC; i += bdim) {
    int dn = i / HC, hc = i % HC, h = hc / C;
    float acc = 0.f;
    for (int sn = 0; sn < kNPG; ++sn) acc += salpha[dn * 64 + h * 32 + sn] * ssrc[sn * HC + hc];
    out[(size_t)(g * kNPG + dn) * ldo + col0 + hc] = acc + bias[hc];
  }
}

// ---------------- edge sum per graph ----------------
__global__ void k_edge_sum(const float* __restrict__ ea, float* __restrict__ esum) {
  int g = blockIdx.x, f = threadIdx.x;  // blockDim = 64
  float acc = 0.f;
  for (int e = 0; e < kEPG; ++e) acc += ea[(size_t)(g * kEPG + e) * kDE + f];
  esum[g * kDE + f] = acc;
}

// ---------------- SAG readout (GraphConv score -> softmax -> weighted sum) ------
__global__ void k_sag(const float* __restrict__ rep, const int* __restrict__ srcI,
                      const int* __restrict__ dstI, const float* __restrict__ W1,
                      const float* __restrict__ W2, const float* __restrict__ pb,
                      const float* __restrict__ eemb, float* __restrict__ embs, int blk) {
  __shared__ float srep[kNPG * kD];
  __shared__ float xd1[kNPG], xd2[kNPG], ssc[kNPG];
  __shared__ int ssl[kEPG], sdl[kEPG];
  __shared__ float sm_, ssum_;
  int g = blockIdx.x, tid = threadIdx.x;  // blockDim = 128
  for (int i = tid; i < kNPG * kD; i += 128) srep[i] = rep[(size_t)g * kNPG * kD + i];
  if (tid < kEPG) {
    ssl[tid] = srcI[g * kEPG + tid] - g * kNPG;
    sdl[tid] = dstI[g * kEPG + tid] - g * kNPG;
  }
  __syncthreads();
  if (tid < kNPG) {
    float d1 = 0.f, d2 = 0.f;
    for (int f = 0; f < kD; ++f) { float v = srep[tid * kD + f]; d1 += v * W1[f]; d2 += v * W2[f]; }
    xd1[tid] = d1; xd2[tid] = d2;
  }
  __syncthreads();
  if (tid < kNPG) {
    float sc = xd1[tid] + pb[0];
    for (int e = 0; e < kEPG; ++e) if (sdl[e] == tid) sc += xd2[ssl[e]];
    ssc[tid] = sc;
  }
  __syncthreads();
  if (tid == 0) {
    float m = -1e30f;
    for (int n = 0; n < kNPG; ++n) m = fmaxf(m, ssc[n]);
    float s = 0.f;
    for (int n = 0; n < kNPG; ++n) s += expf(ssc[n] - m);
    sm_ = m; ssum_ = s;
  }
  __syncthreads();
  float m = sm_, s = ssum_;
  int f = tid;
  float acc = 0.f;
  for (int n = 0; n < kNPG; ++n)
    acc += (expf(ssc[n] - m) / (s + 1e-16f)) * srep[n * kD + f];
  embs[(size_t)(g * kNBLK + blk) * kD + f] = acc + eemb[(size_t)g * kD + f];
}

// ---------------- co-attention tanh scores ----------------
__global__ void k_coattn(const float* __restrict__ Q, const float* __restrict__ Kk,
                         const float* __restrict__ cb, const float* __restrict__ ca,
                         float* __restrict__ attn) {
  __shared__ float sq[kNBLK * 64], sk[kNBLK * 64], red[64];
  int b = blockIdx.x, tid = threadIdx.x;  // blockDim = 64
  for (int i = tid; i < kNBLK * 64; i += 64) {
    sq[i] = Q[(size_t)b * kNBLK * 64 + i];
    sk[i] = Kk[(size_t)b * kNBLK * 64 + i];
  }
  __syncthreads();
  for (int p = 0; p < 16; ++p) {
    int i = p >> 2, j = p & 3;    // i = key(head) block, j = query(tail) block
    red[tid] = tanhf(sq[j * 64 + tid] + sk[i * 64 + tid] + cb[tid]) * ca[tid];
    __syncthreads();
    for (int st = 32; st > 0; st >>= 1) { if (tid < st) red[tid] += red[tid + st]; __syncthreads(); }
    if (tid == 0) attn[b * 16 + p] = red[0];
    __syncthreads();
  }
}

// ---------------- fused L2-normalize + RESCAL + combine ----------------
__global__ void k_final(const float* __restrict__ embsH, const float* __restrict__ embsT,
                        const float* __restrict__ relEmb, const int* __restrict__ rels,
                        const float* __restrict__ attn, float* __restrict__ outp) {
  __shared__ float sh[kNBLK * kD], st[kNBLK * kD], mt[kNBLK * kD], red[128];
  __shared__ float snorm[8];
  int b = blockIdx.x, tid = threadIdx.x;  // blockDim = 128
  for (int i = tid; i < kNBLK * kD; i += 128) {
    sh[i] = embsH[(size_t)b * kNBLK * kD + i];
    st[i] = embsT[(size_t)b * kNBLK * kD + i];
  }
  __syncthreads();
  if (tid < 8) {
    const float* v = (tid < 4) ? (sh + tid * kD) : (st + (tid - 4) * kD);
    float q = 0.f;
    for (int f = 0; f < kD; ++f) q += v[f] * v[f];
    snorm[tid] = fmaxf(sqrtf(q), 1e-12f);
  }
  __syncthreads();
  for (int i = tid; i < kNBLK * kD; i += 128) {
    sh[i] /= snorm[i / kD];
    st[i] /= snorm[4 + i / kD];
  }
  __syncthreads();
  const float* M = relEmb + (size_t)rels[b] * kD * kD;
  for (int j = 0; j < kNBLK; ++j) {   // mt[j][f] = sum_g M[f,g] * tn[j,g]
    int f = tid;
    float acc = 0.f;
    for (int gI = 0; gI < kD; ++gI) acc += M[(size_t)f * kD + gI] * st[j * kD + gI];
    mt[j * kD + f] = acc;
  }
  __syncthreads();
  float res = 0.f;
  for (int p = 0; p < 16; ++p) {
    int i = p >> 2, j = p & 3;
    red[tid] = sh[i * kD + tid] * mt[j * kD + tid];
    __syncthreads();
    for (int stn = 64; stn > 0; stn >>= 1) { if (tid < stn) red[tid] += red[tid + stn]; __syncthreads(); }
    if (tid == 0) res += attn[b * 16 + p] * red[0];
    __syncthreads();
  }
  if (tid == 0) outp[b] = res;
}

// ---------------- host side ----------------
static inline void gemm(const float* A, const float* W, const float* bias, float* C,
                        int M, int K, int Nc, int act, hipStream_t s) {
  dim3 grid(Nc / 64, M / 128);   // all call sites: M multiple of 128, Nc of 64
  if (K == 128)
    k_gemm_wmma<128><<<grid, 256, 0, s>>>(A, W, bias, C, M, Nc, act);
  else
    k_gemm_wmma<64><<<grid, 256, 0, s>>>(A, W, bias, C, M, Nc, act);
}

extern "C" void kernel_launch(void* const* d_in, const int* in_sizes, int n_in,
                              void* d_out, int out_size, void* d_ws, size_t ws_size,
                              hipStream_t stream) {
  (void)in_sizes; (void)n_in; (void)out_size; (void)ws_size;
  const float* h_x  = (const float*)d_in[0];
  const int*   hEI  = (const int*)d_in[1];
  const float* h_ea = (const float*)d_in[2];
  const float* t_x  = (const float*)d_in[5];
  const int*   tEI  = (const int*)d_in[6];
  const float* t_ea = (const float*)d_in[7];
  const int*   rels = (const int*)d_in[11];
  // params, flattened in _make_params insertion order starting at index 12
  const float* Wn   = (const float*)d_in[12];
  const float* bn   = (const float*)d_in[13];
  const float* We   = (const float*)d_in[14];
  const float* be   = (const float*)d_in[15];
  const float* inw  = (const float*)d_in[16];
  const float* inb  = (const float*)d_in[17];
  const float* convW  = (const float*)d_in[18];
  const float* convWe = (const float*)d_in[19];
  const float* cAS  = (const float*)d_in[20];
  const float* cAD  = (const float*)d_in[21];
  const float* cAE  = (const float*)d_in[22];
  const float* cB   = (const float*)d_in[23];
  const float* linW = (const float*)d_in[24];
  const float* linB = (const float*)d_in[25];
  const float* intW = (const float*)d_in[26];
  const float* iAS  = (const float*)d_in[27];
  const float* iAD  = (const float*)d_in[28];
  const float* iB   = (const float*)d_in[29];
  const float* xWs  = (const float*)d_in[30];
  const float* xWd  = (const float*)d_in[31];
  const float* xAS  = (const float*)d_in[32];
  const float* xAD  = (const float*)d_in[33];
  const float* xB   = (const float*)d_in[34];
  const float* pW1  = (const float*)d_in[35];
  const float* pW2  = (const float*)d_in[36];
  const float* pB   = (const float*)d_in[37];
  const float* nW   = (const float*)d_in[38];
  const float* nBb  = (const float*)d_in[39];
  const float* rW   = (const float*)d_in[40];
  const float* rB   = (const float*)d_in[41];
  const float* coWq = (const float*)d_in[42];
  const float* coWk = (const float*)d_in[43];
  const float* coBi = (const float*)d_in[44];
  const float* coA  = (const float*)d_in[45];
  const float* relE = (const float*)d_in[46];
  float* out = (float*)d_out;

  const int* hSrc = hEI;            const int* hDst = hEI + kE;
  const int* tSrc = tEI;            const int* tDst = tEI + kE;

  // workspace layout
  char* ws = (char*)d_ws;
  size_t off = 0;
  auto alloc = [&](size_t nfloats) {
    float* p = (float*)(ws + off);
    off += nfloats * sizeof(float);
    return p;
  };
  float* HX   = alloc((size_t)kN * kD);
  float* TX   = alloc((size_t)kN * kD);
  float* HXE  = alloc((size_t)kN * kD);
  float* TXE  = alloc((size_t)kN * kD);
  float* EAH  = alloc((size_t)kE * kDE);
  float* EAT  = alloc((size_t)kE * kDE);
  float* EATH = alloc((size_t)kE * kDE);   // lin_up output (pre-relu)
  float* EATT = alloc((size_t)kE * kDE);
  float* X1H  = alloc((size_t)kN * kD);    // hs_conv / rep_h / init tmp
  float* X1T  = alloc((size_t)kN * kD);
  float* X2H  = alloc((size_t)kE * kD);    // ew OR [hs_intra | q | kd]
  float* X2T  = alloc((size_t)kE * kD);
  float* EMBH = alloc((size_t)kB * kNBLK * kD);
  float* EMBT = alloc((size_t)kB * kNBLK * kD);
  float* ESH  = alloc((size_t)kB * kDE);
  float* EST  = alloc((size_t)kB * kDE);
  float* EEH  = alloc((size_t)kB * kD);
  float* EET  = alloc((size_t)kB * kD);
  float* KEYS = alloc((size_t)kB * kNBLK * 64);
  float* QUER = alloc((size_t)kB * kNBLK * 64);
  float* ATTN = alloc((size_t)kB * 16);

  float* hsIH = X2H;                         // [N,64]
  float* QH   = X2H + (size_t)kN * 64;       // [N,64]
  float* KDH  = X2H + (size_t)2 * kN * 64;   // [N,64]
  float* hsIT = X2T;
  float* QT   = X2T + (size_t)kN * 64;
  float* KDT  = X2T + (size_t)2 * kN * 64;

  // ---- init: node proj + graph LN + relu; edge proj + relu ----
  k_node_proj<<<kN, kD, 0, stream>>>(h_x, Wn, bn, X1H);
  k_graph_ln<<<kB, 256, 0, stream>>>(X1H, HX, inw, inb, 1);
  k_node_proj<<<kN, kD, 0, stream>>>(t_x, Wn, bn, X1T);
  k_graph_ln<<<kB, 256, 0, stream>>>(X1T, TX, inw, inb, 1);
  k_edge_proj<<<kE, kDE, 0, stream>>>(h_ea, We, be, EAH);
  k_edge_proj<<<kE, kDE, 0, stream>>>(t_ea, We, be, EAT);

  for (int i = 0; i < kNBLK; ++i) {
    const float* convW_i  = convW  + (size_t)i * kD * kD;
    const float* convWe_i = convWe + (size_t)i * kDE * kD;
    const float* cAS_i = cAS + i * 128; const float* cAD_i = cAD + i * 128;
    const float* cAE_i = cAE + i * 128; const float* cB_i  = cB  + i * 128;
    const float* linW_i = linW + (size_t)i * kDE * kDE; const float* linB_i = linB + i * kDE;
    const float* intW_i = intW + (size_t)i * kD * 64;
    const float* iAS_i = iAS + i * 64; const float* iAD_i = iAD + i * 64; const float* iB_i = iB + i * 64;
    const float* xWs_i = xWs + (size_t)i * kD * 64; const float* xWd_i = xWd + (size_t)i * kD * 64;
    const float* xAS_i = xAS + i * 64; const float* xAD_i = xAD + i * 64; const float* xB_i = xB + i * 64;
    const float* pW1_i = pW1 + i * kD; const float* pW2_i = pW2 + i * kD; const float* pB_i = pB + i;
    const float* nW_i = nW + i * kD; const float* nB_i = nBb + i * kD;
    const float* rW_i = rW + (size_t)i * kDE * kD; const float* rB_i = rB + i * kD;

    // conv GAT (with edge features)
    gemm(HX, convW_i, nullptr, X1H, kN, kD, kD, 0, stream);
    gemm(TX, convW_i, nullptr, X1T, kN, kD, kD, 0, stream);
    gemm(EAH, convWe_i, nullptr, X2H, kE, kDE, kD, 0, stream);
    gemm(EAT, convWe_i, nullptr, X2T, kE, kDE, kD, 0, stream);
    k_gat_graph<<<kB, 128, 0, stream>>>(X1H, X2H, cAS_i, cAD_i, cAE_i, hSrc, hDst, cB_i, HX, kD, 0, 2, 64);
    k_gat_graph<<<kB, 128, 0, stream>>>(X1T, X2T, cAS_i, cAD_i, cAE_i, tSrc, tDst, cB_i, TX, kD, 0, 2, 64);

    // edge update (pre-relu stored, relu at block end)
    gemm(EAH, linW_i, linB_i, EATH, kE, kDE, kDE, 0, stream);
    gemm(EAT, linW_i, linB_i, EATT, kE, kDE, kDE, 0, stream);

    // elu
    k_elu<<<(kN * kD + 255) / 256, 256, 0, stream>>>(HX, HXE, kN * kD);
    k_elu<<<(kN * kD + 255) / 256, 256, 0, stream>>>(TX, TXE, kN * kD);

    // intra GAT -> rep cols 0..63
    gemm(HXE, intW_i, nullptr, hsIH, kN, kD, 64, 0, stream);
    gemm(TXE, intW_i, nullptr, hsIT, kN, kD, 64, 0, stream);
    k_gat_graph<<<kB, 128, 0, stream>>>(hsIH, nullptr, iAS_i, iAD_i, nullptr, hSrc, hDst, iB_i, X1H, kD, 0, 2, 32);
    k_gat_graph<<<kB, 128, 0, stream>>>(hsIT, nullptr, iAS_i, iAD_i, nullptr, tSrc, tDst, iB_i, X1T, kD, 0, 2, 32);

    // inter GAT (dense bipartite) -> rep cols 64..127
    gemm(HXE, xWs_i, nullptr, QH, kN, kD, 64, 0, stream);
    gemm(TXE, xWs_i, nullptr, QT, kN, kD, 64, 0, stream);
    gemm(HXE, xWd_i, nullptr, KDH, kN, kD, 64, 0, stream);
    gemm(TXE, xWd_i, nullptr, KDT, kN, kD, 64, 0, stream);
    k_inter_gat<<<dim3(kB, 2), 128, 0, stream>>>(QH, KDT, QT, KDH, xAS_i, xAD_i, xB_i, X1H, X1T, kD, 64);

    // readout: edge-sum -> reshape(relu) -> SAG + add
    k_edge_sum<<<kB, kDE, 0, stream>>>(EATH, ESH);
    k_edge_sum<<<kB, kDE, 0, stream>>>(EATT, EST);
    gemm(ESH, rW_i, rB_i, EEH, kB, kDE, kD, 1, stream);
    gemm(EST, rW_i, rB_i, EET, kB, kDE, kD, 1, stream);
    k_sag<<<kB, 128, 0, stream>>>(X1H, hSrc, hDst, pW1_i, pW2_i, pB_i, EEH, EMBH, i);
    k_sag<<<kB, 128, 0, stream>>>(X1T, tSrc, tDst, pW1_i, pW2_i, pB_i, EET, EMBT, i);

    // block end: graph LN + relu on reps; relu edge feats
    k_graph_ln<<<kB, 256, 0, stream>>>(X1H, HX, nW_i, nB_i, 1);
    k_graph_ln<<<kB, 256, 0, stream>>>(X1T, TX, nW_i, nB_i, 1);
    k_relu<<<(kE * kDE + 255) / 256, 256, 0, stream>>>(EATH, EAH, kE * kDE);
    k_relu<<<(kE * kDE + 255) / 256, 256, 0, stream>>>(EATT, EAT, kE * kDE);
  }

  // co-attention + RESCAL
  gemm(EMBH, coWk, nullptr, KEYS, kB * kNBLK, kD, 64, 0, stream);  // keys from heads
  gemm(EMBT, coWq, nullptr, QUER, kB * kNBLK, kD, 64, 0, stream);  // queries from tails
  k_coattn<<<kB, 64, 0, stream>>>(QUER, KEYS, coBi, coA, ATTN);
  k_final<<<kB, 128, 0, stream>>>(EMBH, EMBT, relE, rels, ATTN, out);
}
